// FlashAttention_10307921510925
// MI455X (gfx1250) — matrix operand order repaired and checked by reading, not compiled, not run on hardware
//
#include <hip/hip_runtime.h>

typedef __attribute__((ext_vector_type(16))) __bf16 v16bf;
typedef __attribute__((ext_vector_type(8)))  float  v8f;

#define DIMC   1024
#define NHEADS 16
#define HDIM   64
#define SEQ    2048
#define BATCH  2
#define ROWS   (BATCH*SEQ)   /* 4096 */
#define J3     (3*DIMC)      /* 3072 */

static __device__ __forceinline__ unsigned short f2bf(float f) {
  unsigned u = __float_as_uint(f);
  u += 0x7FFFu + ((u >> 16) & 1u);   // round-to-nearest-even
  return (unsigned short)(u >> 16);
}

static __device__ __forceinline__ v8f vzero() {
  v8f z = {0.f, 0.f, 0.f, 0.f, 0.f, 0.f, 0.f, 0.f};
  return z;
}

// A-operand fragment (16x32 bf16). Lane holds row M=lane&15.
// halves e=0..7 -> K = k0+hi*8+e ; e=8..15 -> K = k0+16+hi*8+(e-8)
static __device__ __forceinline__ v16bf load_fragA(const unsigned short* row, int k0, int hi) {
  uint4 lo = *(const uint4*)(row + k0 + hi * 8);
  uint4 hh = *(const uint4*)(row + k0 + 16 + hi * 8);
  v16bf r;
  __builtin_memcpy(&r, &lo, 16);
  __builtin_memcpy((char*)&r + 16, &hh, 16);
  return r;
}

// B-operand fragment (32x16 bf16) read from BT (N rows x K cols row-major).
// Lane holds column N=lane&15; halves e=0..7 -> K = k0+hi*8+e ; e=8..15 -> K = k0+16+hi*8+(e-8)
static __device__ __forceinline__ v16bf load_fragB(const unsigned short* rowT, int k0, int hi) {
  uint4 lo = *(const uint4*)(rowT + k0 + hi * 8);
  uint4 hh = *(const uint4*)(rowT + k0 + 16 + hi * 8);
  v16bf r;
  __builtin_memcpy(&r, &lo, 16);
  __builtin_memcpy((char*)&r + 16, &hh, 16);
  return r;
}

#define WMMA_BF16(a, b, c) \
  __builtin_amdgcn_wmma_f32_16x16x32_bf16(false, (a), false, (b), (short)0, (c), false, false)

// ---------------- conversion kernels ----------------

__global__ void __launch_bounds__(256) k_cvt(const float* __restrict__ in,
                                             unsigned short* __restrict__ out, int n) {
  for (int i = blockIdx.x * blockDim.x + threadIdx.x; i < n; i += gridDim.x * blockDim.x)
    out[i] = f2bf(in[i]);
}

__global__ void __launch_bounds__(256) k_transpose(const float* __restrict__ in,
                                                   unsigned short* __restrict__ out,
                                                   int rows, int cols) {
  int n = rows * cols;
  for (int i = blockIdx.x * blockDim.x + threadIdx.x; i < n; i += gridDim.x * blockDim.x) {
    int r = i / cols, c = i - r * cols;
    out[(size_t)c * rows + r] = f2bf(in[i]);
  }
}

// ---------------- QKV GEMM: qkv = x @ w_qkv + b, scatter to Q/K/V^T (bf16) ----------------

__global__ void __launch_bounds__(128) k_qkv(
    const unsigned short* __restrict__ xb, const unsigned short* __restrict__ wT,
    const float* __restrict__ bias, unsigned short* __restrict__ Qd,
    unsigned short* __restrict__ Kd, unsigned short* __restrict__ VT) {
  const int lane = threadIdx.x & 31, hi = lane >> 4, col = lane & 15;
  const int wid = threadIdx.x >> 5;
  const int tile = blockIdx.x * 4 + wid;
  const int JT = J3 / 64;                       // 48
  const int mt = tile / JT, jt = tile - mt * JT;
  const int mbase = mt * 64, jbase = jt * 64;

  const unsigned short* arow[4];
  const unsigned short* brow[4];
#pragma unroll
  for (int i = 0; i < 4; ++i) {
    arow[i] = xb + (size_t)(mbase + i * 16 + col) * DIMC;
    brow[i] = wT + (size_t)(jbase + i * 16 + col) * DIMC;
  }
  v8f acc[4][4];
#pragma unroll
  for (int mi = 0; mi < 4; ++mi)
#pragma unroll
    for (int ni = 0; ni < 4; ++ni) acc[mi][ni] = vzero();

  for (int kk = 0; kk < DIMC; kk += 32) {
    v16bf a[4], b[4];
#pragma unroll
    for (int i = 0; i < 4; ++i) {
      a[i] = load_fragA(arow[i], kk, hi);
      b[i] = load_fragB(brow[i], kk, hi);
    }
#pragma unroll
    for (int mi = 0; mi < 4; ++mi)
#pragma unroll
      for (int ni = 0; ni < 4; ++ni)
        acc[mi][ni] = WMMA_BF16(a[mi], b[ni], acc[mi][ni]);
  }

#pragma unroll
  for (int ni = 0; ni < 4; ++ni) {
    const int jcol0 = jbase + ni * 16;
    const int j = jcol0 + col;
    const float bv = bias[j];
    if (jcol0 < DIMC) {                         // ---- Q region (scaled by D^-0.5) ----
      const int h = j >> 6, d = j & 63;
#pragma unroll
      for (int mi = 0; mi < 4; ++mi) {
        const int m0 = mbase + mi * 16 + hi * 8;
        const int b_ = m0 >> 11, n0 = m0 & (SEQ - 1);
        unsigned short* dst = Qd + ((size_t)(b_ * NHEADS + h) * SEQ + n0) * HDIM + d;
#pragma unroll
        for (int g = 0; g < 8; ++g)
          dst[(size_t)g * HDIM] = f2bf((acc[mi][ni][g] + bv) * 0.125f);
      }
    } else if (jcol0 < 2 * DIMC) {              // ---- K region ----
      const int jj = j - DIMC;
      const int h = jj >> 6, d = jj & 63;
#pragma unroll
      for (int mi = 0; mi < 4; ++mi) {
        const int m0 = mbase + mi * 16 + hi * 8;
        const int b_ = m0 >> 11, n0 = m0 & (SEQ - 1);
        unsigned short* dst = Kd + ((size_t)(b_ * NHEADS + h) * SEQ + n0) * HDIM + d;
#pragma unroll
        for (int g = 0; g < 8; ++g)
          dst[(size_t)g * HDIM] = f2bf(acc[mi][ni][g] + bv);
      }
    } else {                                    // ---- V region -> V^T [b][h][d][n] ----
      const int jj = j - 2 * DIMC;
      const int h = jj >> 6, d = jj & 63;
#pragma unroll
      for (int mi = 0; mi < 4; ++mi) {
        const int m0 = mbase + mi * 16 + hi * 8;
        const int b_ = m0 >> 11, n0 = m0 & (SEQ - 1);
        uint4 pk;
        unsigned short* ps = (unsigned short*)&pk;
#pragma unroll
        for (int g = 0; g < 8; ++g) ps[g] = f2bf(acc[mi][ni][g] + bv);
        *(uint4*)(VT + ((size_t)(b_ * NHEADS + h) * HDIM + d) * SEQ + n0) = pk;
      }
    }
  }
}

// ---------------- Flash attention: one wave = 16 queries x 2048 keys ----------------

__global__ void __launch_bounds__(128) k_attn(
    const unsigned short* __restrict__ Qd, const unsigned short* __restrict__ Kd,
    const unsigned short* __restrict__ VT, unsigned short* __restrict__ AO) {
  const int lane = threadIdx.x & 31, hi = lane >> 4, col = lane & 15;
  const int wid = threadIdx.x >> 5;
  const int t = blockIdx.x * 4 + wid;
  const int bh = t >> 7;                        // 128 query tiles per (b,h)
  const int qbase = (t & 127) * 16;

  const unsigned short* Qh = Qd + (size_t)bh * SEQ * HDIM;
  const unsigned short* Kh = Kd + (size_t)bh * SEQ * HDIM;
  const unsigned short* Vh = VT + (size_t)bh * HDIM * SEQ;

  // loop-invariant q B-operands (K = head dim, split 0..31 / 32..63)
  const unsigned short* qrow = Qh + (size_t)(qbase + col) * HDIM;
  const v16bf qb0 = load_fragB(qrow, 0, hi);
  const v16bf qb1 = load_fragB(qrow, 32, hi);

  v8f o[4];
#pragma unroll
  for (int di = 0; di < 4; ++di) o[di] = vzero();
  float mrun = -__builtin_inff();
  float lrun = 0.f;

  for (int kb = 0; kb < SEQ; kb += 32) {
    // S^T = K_tile @ q^T : lane owns query column col, rows = keys
    const unsigned short* k0row = Kh + (size_t)(kb + col) * HDIM;
    const unsigned short* k1row = k0row + 16 * HDIM;
    v8f s0 = vzero(), s1 = vzero();
    s0 = WMMA_BF16(load_fragA(k0row, 0, hi), qb0, s0);
    s0 = WMMA_BF16(load_fragA(k0row, 32, hi), qb1, s0);
    s1 = WMMA_BF16(load_fragA(k1row, 0, hi), qb0, s1);
    s1 = WMMA_BF16(load_fragA(k1row, 32, hi), qb1, s1);

    // online softmax: per-query state lives redundantly in lanes col and col+16
    float smax = -__builtin_inff();
#pragma unroll
    for (int g = 0; g < 8; ++g) smax = fmaxf(smax, fmaxf(s0[g], s1[g]));
    smax = fmaxf(smax, __shfl_xor(smax, 16, 32));
    const float mnew = fmaxf(mrun, smax);
    const float corr = __expf(mrun - mnew);     // first chunk: exp(-inf)=0
    float lsum = 0.f;
    unsigned short ps[16];
#pragma unroll
    for (int g = 0; g < 8; ++g) {
      float p = __expf(s0[g] - mnew);
      lsum += p; ps[g] = f2bf(p);
    }
#pragma unroll
    for (int g = 0; g < 8; ++g) {
      float p = __expf(s1[g] - mnew);
      lsum += p; ps[8 + g] = f2bf(p);
    }
    lsum += __shfl_xor(lsum, 16, 32);
    lrun = lrun * corr + lsum;
    mrun = mnew;

    // rescale O accumulators; query of O-row g is (hi*8+g), state held in lane (hi*8+g)
#pragma unroll
    for (int g = 0; g < 8; ++g) {
      const float cg = __shfl(corr, hi * 8 + g, 32);
#pragma unroll
      for (int di = 0; di < 4; ++di) o[di][g] *= cg;
    }

    // S^T C-fragment key set per lane == A-fragment k-map  ->  P repacks in-lane as A
    v16bf pa;
    __builtin_memcpy(&pa, ps, 32);
#pragma unroll
    for (int di = 0; di < 4; ++di) {
      const unsigned short* vrow = Vh + (size_t)(di * 16 + col) * SEQ;  // V^T row d
      o[di] = WMMA_BF16(pa, load_fragB(vrow, kb, hi), o[di]);
    }
  }

  const float linv = 1.f / lrun;
  const int b_ = bh >> 4, h = bh & 15;
#pragma unroll
  for (int g = 0; g < 8; ++g) {
    const float lg = __shfl(linv, hi * 8 + g, 32);
    const int n = qbase + hi * 8 + g;
    unsigned short* dst = AO + ((size_t)(b_ * SEQ + n)) * DIMC + h * HDIM + col;
#pragma unroll
    for (int di = 0; di < 4; ++di) dst[di * 16] = f2bf(o[di][g] * lg);
  }
}

// ---------------- Output projection: out = attn @ w_proj + b (fp32 out) ----------------

__global__ void __launch_bounds__(128) k_proj(
    const unsigned short* __restrict__ A, const unsigned short* __restrict__ wT,
    const float* __restrict__ bias, float* __restrict__ out) {
  const int lane = threadIdx.x & 31, hi = lane >> 4, col = lane & 15;
  const int wid = threadIdx.x >> 5;
  const int tile = blockIdx.x * 4 + wid;
  const int JT = DIMC / 64;                     // 16
  const int mt = tile / JT, jt = tile - mt * JT;
  const int mbase = mt * 64, jbase = jt * 64;

  const unsigned short* arow[4];
  const unsigned short* brow[4];
#pragma unroll
  for (int i = 0; i < 4; ++i) {
    arow[i] = A + (size_t)(mbase + i * 16 + col) * DIMC;
    brow[i] = wT + (size_t)(jbase + i * 16 + col) * DIMC;
  }
  v8f acc[4][4];
#pragma unroll
  for (int mi = 0; mi < 4; ++mi)
#pragma unroll
    for (int ni = 0; ni < 4; ++ni) acc[mi][ni] = vzero();

  for (int kk = 0; kk < DIMC; kk += 32) {
    v16bf a[4], b[4];
#pragma unroll
    for (int i = 0; i < 4; ++i) {
      a[i] = load_fragA(arow[i], kk, hi);
      b[i] = load_fragB(brow[i], kk, hi);
    }
#pragma unroll
    for (int mi = 0; mi < 4; ++mi)
#pragma unroll
      for (int ni = 0; ni < 4; ++ni)
        acc[mi][ni] = WMMA_BF16(a[mi], b[ni], acc[mi][ni]);
  }

#pragma unroll
  for (int ni = 0; ni < 4; ++ni) {
    const int j = jbase + ni * 16 + col;
    const float bv = bias[j];
#pragma unroll
    for (int mi = 0; mi < 4; ++mi) {
      const int m0 = mbase + mi * 16 + hi * 8;
#pragma unroll
      for (int g = 0; g < 8; ++g)
        out[(size_t)(m0 + g) * DIMC + j] = acc[mi][ni][g] + bv;
    }
  }
}

// ---------------- launch ----------------

extern "C" void kernel_launch(void* const* d_in, const int* in_sizes, int n_in,
                              void* d_out, int out_size, void* d_ws, size_t ws_size,
                              hipStream_t stream) {
  const float* x      = (const float*)d_in[0];
  const float* w_qkv  = (const float*)d_in[1];
  const float* b_qkv  = (const float*)d_in[2];
  const float* w_proj = (const float*)d_in[3];
  const float* b_proj = (const float*)d_in[4];
  float* out = (float*)d_out;

  // workspace layout (bf16 halves), total 48 MB
  unsigned short* xb     = (unsigned short*)d_ws;              // 4096*1024
  unsigned short* wqkvT  = xb + (size_t)ROWS * DIMC;           // 3072*1024
  unsigned short* wprojT = wqkvT + (size_t)J3 * DIMC;          // 1024*1024
  unsigned short* Qd     = wprojT + (size_t)DIMC * DIMC;       // 2*16*2048*64
  unsigned short* Kd     = Qd + (size_t)BATCH * NHEADS * SEQ * HDIM;
  unsigned short* VT     = Kd + (size_t)BATCH * NHEADS * SEQ * HDIM;
  unsigned short* AO     = VT + (size_t)BATCH * NHEADS * SEQ * HDIM;  // 4096*1024

  k_cvt<<<2048, 256, 0, stream>>>(x, xb, ROWS * DIMC);
  k_transpose<<<2048, 256, 0, stream>>>(w_qkv, wqkvT, DIMC, J3);
  k_transpose<<<1024, 256, 0, stream>>>(w_proj, wprojT, DIMC, DIMC);

  // (4096/64) * (3072/64) = 3072 wave-tiles, 4 waves/block
  k_qkv<<<768, 128, 0, stream>>>(xb, wqkvT, b_qkv, Qd, Kd, VT);
  // 32 heads * 128 query tiles = 4096 wave-tiles
  k_attn<<<1024, 128, 0, stream>>>(Qd, Kd, VT, AO);
  // (4096/64) * (1024/64) = 1024 wave-tiles
  k_proj<<<256, 128, 0, stream>>>(AO, wprojT, b_proj, out);
}
